// MoE_9835475107967
// MI455X (gfx1250) — matrix operand order repaired and checked by reading, not compiled, not run on hardware
//
#include <hip/hip_runtime.h>

typedef __attribute__((ext_vector_type(16))) __bf16 v16bf;
typedef __attribute__((ext_vector_type(8)))  float  v8f;

#define HDIM 1024
#define NTOK 1024
#define NEXP 8

// pack two f32 -> bf16x2 dword via one v_perm_b32 (truncating conversion)
__device__ __forceinline__ unsigned pack2bf(float lo, float hi) {
#if __has_builtin(__builtin_amdgcn_perm)
  return __builtin_amdgcn_perm(__builtin_bit_cast(unsigned, hi),
                               __builtin_bit_cast(unsigned, lo), 0x07060302u);
#else
  return (__builtin_bit_cast(unsigned, lo) >> 16) |
         (__builtin_bit_cast(unsigned, hi) & 0xFFFF0000u);
#endif
}

// truncating f32 -> bf16 (maps to ds_store_b16_d16_hi when stored)
__device__ __forceinline__ unsigned short f2bfu(float f) {
  return (unsigned short)(__builtin_bit_cast(unsigned, f) >> 16);
}

__device__ __forceinline__ float fast_rcp(float x) {
#if __has_builtin(__builtin_amdgcn_rcpf)
  return __builtin_amdgcn_rcpf(x);
#else
  return 1.0f / x;
#endif
}

union BF16x16 {            // one WMMA A/B fragment, buildable as 8 dwords
  v16bf v;
  unsigned u[8];
  uint4 q[2];
};

__global__ void zero_kernel(float* __restrict__ p, int n) {
  int i = blockIdx.x * blockDim.x + threadIdx.x;
  if (i < n) p[i] = 0.0f;
}

// ---- gating: softmax(x @ router_w) -> top-2 scores/indices per token ----
__global__ void router_kernel(const float* __restrict__ x,
                              const float* __restrict__ rw,
                              int* __restrict__ tidx, float* __restrict__ tw) {
  int n = blockIdx.x * blockDim.x + threadIdx.x;
  if (n >= NTOK) return;
  const float* xr = x + (size_t)n * HDIM;
  float acc[NEXP];
  #pragma unroll
  for (int e = 0; e < NEXP; ++e) acc[e] = 0.0f;
  for (int h = 0; h < HDIM; ++h) {
    float xv = xr[h];
    #pragma unroll
    for (int e = 0; e < NEXP; ++e) acc[e] += xv * rw[h * NEXP + e];
  }
  float m = acc[0];
  #pragma unroll
  for (int e = 1; e < NEXP; ++e) m = fmaxf(m, acc[e]);
  float s = 0.0f;
  #pragma unroll
  for (int e = 0; e < NEXP; ++e) { acc[e] = __expf(acc[e] - m); s += acc[e]; }
  float inv = 1.0f / s;
  #pragma unroll
  for (int e = 0; e < NEXP; ++e) acc[e] *= inv;
  int b0 = 0; float v0 = acc[0];
  #pragma unroll
  for (int e = 1; e < NEXP; ++e) if (acc[e] > v0) { v0 = acc[e]; b0 = e; }
  int b1 = 0; float v1 = -1.0f;
  #pragma unroll
  for (int e = 0; e < NEXP; ++e) if (e != b0 && acc[e] > v1) { v1 = acc[e]; b1 = e; }
  tidx[n * 2]     = b0; tidx[n * 2 + 1] = b1;
  tw[n * 2]       = v0; tw[n * 2 + 1]   = v1;
}

// ---- deterministic per-expert token list compaction (1 thread/expert) ----
__global__ void build_kernel(const int* __restrict__ tidx, const float* __restrict__ tw,
                             int* __restrict__ counts, int* __restrict__ etok,
                             float* __restrict__ escl) {
  int e = threadIdx.x;
  if (e >= NEXP) return;
  int c = 0;
  for (int n = 0; n < NTOK; ++n) {
    #pragma unroll
    for (int j = 0; j < 2; ++j) {
      if (tidx[n * 2 + j] == e) {
        etok[e * NTOK + c] = n;
        escl[e * NTOK + c] = tw[n * 2 + j];
        ++c;
      }
    }
  }
  counts[e] = c;
}

// ---- fused SiLU-gated MLP for one (expert, 16-token tile) workgroup ----
// tokens==nullptr -> identity gather / scale 1 (shared expert path)
__global__ __launch_bounds__(128, 1)
void ffn_kernel(const float* __restrict__ X,
                const int* __restrict__ tokens,
                const float* __restrict__ scales,
                const int* __restrict__ counts,
                const float* __restrict__ Wg,
                const float* __restrict__ Wu,
                const float* __restrict__ Wd,
                int Fdim, float* __restrict__ out) {
  const int e   = blockIdx.z;
  const int m0  = blockIdx.x * 16;
  const int cnt = counts ? counts[e] : NTOK;
  if (m0 >= cnt) return;

  // uniform (SGPR) bases; per-lane addressing is 32-bit indices so the
  // backend emits saddr-form global loads with folded immediate offsets.
  const float* wg = Wg + (size_t)e * HDIM * Fdim;
  const float* wu = Wu + (size_t)e * HDIM * Fdim;
  const float* wd = Wd + (size_t)e * Fdim * HDIM;

  __shared__ __align__(16) unsigned short lds_h[16 * 128];  // h tile, bf16
  __shared__ int   lds_tok[16];
  __shared__ float lds_scl[16];

  const int tid  = threadIdx.x;
  const int wave = tid >> 5;
  const int lane = tid & 31;
  const int half = lane >> 4;
  const int l16  = lane & 15;

  if (tid < 16) {
    int m = m0 + tid;
    int tok = 0; float sc = 0.0f;
    if (m < cnt) {
      if (tokens) { tok = tokens[(size_t)e * NTOK + m]; sc = scales[(size_t)e * NTOK + m]; }
      else        { tok = m; sc = 1.0f; }
    }
    lds_tok[tid] = tok;
    lds_scl[tid] = sc;
  }
  __syncthreads();

  const float* xrow = X + (size_t)lds_tok[l16] * HDIM;  // A row for this lane

  // hoist per-row scale / token id (C-layout rows for this lane)
  float sclr[8];
  int   tokr[8];
  #pragma unroll
  for (int r = 0; r < 8; ++r) {
    sclr[r] = lds_scl[r + 8 * half];
    tokr[r] = lds_tok[r + 8 * half];
  }

  const v8f zacc = {0, 0, 0, 0, 0, 0, 0, 0};
  v8f outacc[16];
  #pragma unroll
  for (int i = 0; i < 16; ++i) outacc[i] = zacc;

  const int nft = Fdim >> 7;  // 128-col F tiles
  for (int ft = 0; ft < nft; ++ft) {
    // ---------------- phase 1: g,u = X @ Wg, X @ Wu (this wave: 32 cols) ----
    const int cbase = ft * 128 + wave * 32 + l16;
    v8f g0 = zacc, g1 = zacc, u0 = zacc, u1 = zacc;
    for (int k0 = 0; k0 < HDIM; k0 += 32) {
      // A fragment: 16x32 bf16, row = l16, K = k0 + half*16 + s
      const float* ap = xrow + k0 + half * 16;
      float4 f0 = *(const float4*)(ap);
      float4 f1 = *(const float4*)(ap + 4);
      float4 f2 = *(const float4*)(ap + 8);
      float4 f3 = *(const float4*)(ap + 12);
      BF16x16 a;
      a.u[0] = pack2bf(f0.x, f0.y); a.u[1] = pack2bf(f0.z, f0.w);
      a.u[2] = pack2bf(f1.x, f1.y); a.u[3] = pack2bf(f1.z, f1.w);
      a.u[4] = pack2bf(f2.x, f2.y); a.u[5] = pack2bf(f2.z, f2.w);
      a.u[6] = pack2bf(f3.x, f3.y); a.u[7] = pack2bf(f3.z, f3.w);

      // B fragments: col = l16 (+16 for second subtile), K = k0 + half*16 + s
      const int koff = (k0 + half * 16) * Fdim + cbase;   // 32-bit lane index
      BF16x16 bg0, bg1, bu0, bu1;
      #pragma unroll
      for (int j = 0; j < 8; ++j) {
        int o0 = koff + (2 * j) * Fdim;
        int o1 = koff + (2 * j + 1) * Fdim;
        bg0.u[j] = pack2bf(wg[o0],      wg[o1]);
        bg1.u[j] = pack2bf(wg[o0 + 16], wg[o1 + 16]);
        bu0.u[j] = pack2bf(wu[o0],      wu[o1]);
        bu1.u[j] = pack2bf(wu[o0 + 16], wu[o1 + 16]);
      }
      g0 = __builtin_amdgcn_wmma_f32_16x16x32_bf16(false, a.v, false, bg0.v, (short)0, g0, false, false);
      g1 = __builtin_amdgcn_wmma_f32_16x16x32_bf16(false, a.v, false, bg1.v, (short)0, g1, false, false);
      u0 = __builtin_amdgcn_wmma_f32_16x16x32_bf16(false, a.v, false, bu0.v, (short)0, u0, false, false);
      u1 = __builtin_amdgcn_wmma_f32_16x16x32_bf16(false, a.v, false, bu1.v, (short)0, u1, false, false);
    }

    // ---------------- phase 2: h = silu(g)*u*score -> LDS (bf16) ------------
    __syncthreads();  // previous tile's phase-3 readers are done with lds_h
    #pragma unroll
    for (int r = 0; r < 8; ++r) {
      int row = r + 8 * half;                 // C layout: row = vgpr + 8*(lane/16)
      float sc = sclr[r];
      float gg0 = g0[r], gg1 = g1[r];
      float h0 = gg0 * fast_rcp(1.0f + __expf(-gg0)) * u0[r] * sc;
      float h1 = gg1 * fast_rcp(1.0f + __expf(-gg1)) * u1[r] * sc;
      lds_h[row * 128 + wave * 32 + l16]      = f2bfu(h0);
      lds_h[row * 128 + wave * 32 + 16 + l16] = f2bfu(h1);
    }
    __syncthreads();

    // ---------------- phase 3: outacc += h @ Wd (this wave: 256 H cols) -----
    BF16x16 ah[4];
    const unsigned short* hrow = &lds_h[l16 * 128];
    #pragma unroll
    for (int kk = 0; kk < 4; ++kk) {
      ah[kk].q[0] = *(const uint4*)(hrow + kk * 32 + half * 16);       // slots 0..7
      ah[kk].q[1] = *(const uint4*)(hrow + kk * 32 + half * 16 + 8);   // slots 8..15
    }
    const int fbase = ft * 128;
    #pragma unroll
    for (int nt = 0; nt < 16; ++nt) {
      const int ncol = wave * 256 + nt * 16 + l16;
      v8f c = outacc[nt];
      #pragma unroll
      for (int kk = 0; kk < 4; ++kk) {
        const int doff = (fbase + kk * 32 + half * 16) * HDIM + ncol;  // 32-bit
        BF16x16 b;
        #pragma unroll
        for (int j = 0; j < 8; ++j) {
          b.u[j] = pack2bf(wd[doff + (2 * j) * HDIM], wd[doff + (2 * j + 1) * HDIM]);
        }
        c = __builtin_amdgcn_wmma_f32_16x16x32_bf16(false, ah[kk].v, false, b.v, (short)0, c, false, false);
      }
      outacc[nt] = c;
    }
  }

  // ---------------- epilogue: scatter-add into out -----------------------
  #pragma unroll
  for (int nt = 0; nt < 16; ++nt) {
    int col = wave * 256 + nt * 16 + l16;
    #pragma unroll
    for (int r = 0; r < 8; ++r) {
      if (m0 + (r + 8 * half) < cnt) {
        atomicAdd(&out[(size_t)tokr[r] * HDIM + col], outacc[nt][r]);
      }
    }
  }
}

extern "C" void kernel_launch(void* const* d_in, const int* in_sizes, int n_in,
                              void* d_out, int out_size, void* d_ws, size_t ws_size,
                              hipStream_t stream) {
  (void)in_sizes; (void)n_in; (void)ws_size;
  const float* x   = (const float*)d_in[0];
  const float* rw  = (const float*)d_in[1];
  const float* gw  = (const float*)d_in[2];
  const float* uw  = (const float*)d_in[3];
  const float* dw  = (const float*)d_in[4];
  const float* sgw = (const float*)d_in[5];
  const float* suw = (const float*)d_in[6];
  const float* sdw = (const float*)d_in[7];
  float* out = (float*)d_out;

  char* ws      = (char*)d_ws;
  int*   tidx   = (int*)  (ws);                   // 2048 ints
  float* tw     = (float*)(ws + 8 * 1024);        // 2048 floats
  int*   counts = (int*)  (ws + 16 * 1024);       // 8 ints
  int*   etok   = (int*)  (ws + 16 * 1024 + 256); // 8*1024 ints
  float* escl   = (float*)(ws + 48 * 1024 + 256); // 8*1024 floats

  zero_kernel<<<(out_size + 255) / 256, 256, 0, stream>>>(out, out_size);
  router_kernel<<<NTOK / 256, 256, 0, stream>>>(x, rw, tidx, tw);
  build_kernel<<<1, 32, 0, stream>>>(tidx, tw, counts, etok, escl);

  // shared expert: identity gather, Fs = 2048
  ffn_kernel<<<dim3(64, 1, 1), 128, 0, stream>>>(x, nullptr, nullptr, nullptr,
                                                 sgw, suw, sdw, 2048, out);
  // routed experts: gathered token lists, F = 1024, grid.z = expert
  ffn_kernel<<<dim3(64, 1, NEXP), 128, 0, stream>>>(x, etok, escl, counts,
                                                    gw, uw, dw, 1024, out);
}
